// DecoderLayer_42975442764249
// MI455X (gfx1250) — matrix operand order repaired and checked by reading, not compiled, not run on hardware
//
#include <hip/hip_runtime.h>
#include <math.h>

// ---------------------------------------------------------------------------
// MI455X (gfx1250) decoder layer: bf16 WMMA GEMMs + softmax + global LayerNorm
// Async memory->LDS pipeline (GLOBAL_LOAD_ASYNC_TO_LDS_B128 + s_wait_asynccnt)
// ---------------------------------------------------------------------------

typedef __bf16 bf16_t;
typedef __bf16 v16bf __attribute__((ext_vector_type(16)));
typedef __bf16 v8bf  __attribute__((ext_vector_type(8)));
typedef float  v8f   __attribute__((ext_vector_type(8)));

#define S_DIM  2048
#define E_DIM  2048
#define H_DIM  16
#define KD_DIM 128
#define FF_DIM 8192
#define LN_EPS 1e-5f

#define BM 128
#define BN 128
#define BK 32
#define LDS_PITCH (BK + 8)   // 80B row stride: 16B-aligned chunks, no bank conflicts

enum { EPI_RELU = 1, EPI_BF16 = 2, EPI_CAUSAL = 4 };

// CDNA5 async memory->LDS copies (ASYNCcnt-tracked, no VGPR round trip).
// The load itself is emitted via inline asm (the clang builtin's pointer
// params use CUDA LangAS-qualified types that HIP source cannot spell);
// the ASYNCcnt waits use the verified builtin.
#if defined(__has_builtin)
#if __has_builtin(__builtin_amdgcn_global_load_async_to_lds_b128) && \
    __has_builtin(__builtin_amdgcn_s_wait_asynccnt)
#define USE_ASYNC_LDS 1
#endif
#endif

__device__ __forceinline__ void copy16_to_lds(bf16_t* lds, const bf16_t* g)
{
#if defined(USE_ASYNC_LDS)
    // low 32 bits of a generic LDS pointer are the LDS byte offset
    const unsigned lds_off = (unsigned)(unsigned long long)lds;
    asm volatile("global_load_async_to_lds_b128 %0, %1, off"
                 :: "v"(lds_off), "v"(g)
                 : "memory");
#else
    *(uint4*)lds = *(const uint4*)g;
#endif
}

__device__ __forceinline__ void wait_async_tail4()
{
#if defined(USE_ASYNC_LDS)
    __builtin_amdgcn_s_wait_asynccnt(4);   // retire previous tile's 4 copies
#endif
}
__device__ __forceinline__ void wait_async_all()
{
#if defined(USE_ASYNC_LDS)
    __builtin_amdgcn_s_wait_asynccnt(0);
#endif
}

// ---------------------------------------------------------------------------
// Tiled bf16 WMMA GEMM:  C[z] = alpha * A[z][M,K] * Bt[z][N,K] (+ bias[n])
// A row-major [M,K], Bt is B stored column-major as [N,K] rows.
// "K super-batch": K axis batched in chunks of 2^kpb_shift with extra strides
// (used to express attn_out = sum over heads as a single K=H*S GEMM).
// Double-buffered LDS; tile i+1 streams in (async) while tile i is in WMMA.
// ---------------------------------------------------------------------------
__global__ __launch_bounds__(256, 2)
void gemm_bf16_wmma(const bf16_t* __restrict__ A, const bf16_t* __restrict__ B,
                    void* __restrict__ C, const float* __restrict__ bias,
                    int M, int N, int K,
                    long lda, long ldb, long ldc,
                    long a_zstride, long b_zstride, long c_zstride, long bias_zstride,
                    int kpb_shift, long a_kb_stride, long b_kb_stride,
                    float alpha, int flags)
{
    __shared__ bf16_t sA[2][BM][LDS_PITCH];
    __shared__ bf16_t sB[2][BN][LDS_PITCH];

    const int m0 = blockIdx.y * BM;
    const int n0 = blockIdx.x * BN;
    // causal: blocks strictly above the diagonal are never read by softmax
    if ((flags & EPI_CAUSAL) && (n0 > m0 + BM - 1)) return;

    const int z = blockIdx.z;
    A += (long)z * a_zstride;
    B += (long)z * b_zstride;
    const float* biasz = bias ? (bias + (long)z * bias_zstride) : nullptr;

    const int tid   = threadIdx.x;
    const int lane  = tid & 31;
    const int wave  = tid >> 5;           // 8 waves
    const int wmb   = (wave & 3) * 32;    // wave tile: 32 rows (M)
    const int wnb   = (wave >> 2) * 64;   // wave tile: 64 cols (N)
    const int lrow  = lane & 15;
    const int khalf = lane >> 4;          // half-wave K-group select

    v8f acc[2][4];
    const v8f zacc = {0.f, 0.f, 0.f, 0.f, 0.f, 0.f, 0.f, 0.f};
#pragma unroll
    for (int i = 0; i < 2; i++)
#pragma unroll
        for (int j = 0; j < 4; j++) acc[i][j] = zacc;

    // 512 16-byte chunks per 128x32 bf16 tile; 256 threads -> 2 chunks each
    const int ld_row = tid >> 2;          // 0..63
    const int ld_col = (tid & 3) * 8;     // element offset (8 bf16 = 16B)
    const long arow0 = (long)(m0 + ld_row)      * lda + ld_col;
    const long arow1 = (long)(m0 + ld_row + 64) * lda + ld_col;
    const long brow0 = (long)(n0 + ld_row)      * ldb + ld_col;
    const long brow1 = (long)(n0 + ld_row + 64) * ldb + ld_col;
    const int  kpb_mask = (1 << kpb_shift) - 1;

    auto issue_tile = [&](int kt, int buf) {
        const int hb = kt >> kpb_shift;            // power-of-two super-batch
        const int kk = kt & kpb_mask;
        const bf16_t* Ab = A + (long)hb * a_kb_stride + kk;
        const bf16_t* Bb = B + (long)hb * b_kb_stride + kk;
        copy16_to_lds(&sA[buf][ld_row     ][ld_col], Ab + arow0);
        copy16_to_lds(&sA[buf][ld_row + 64][ld_col], Ab + arow1);
        copy16_to_lds(&sB[buf][ld_row     ][ld_col], Bb + brow0);
        copy16_to_lds(&sB[buf][ld_row + 64][ld_col], Bb + brow1);
#if !defined(USE_ASYNC_LDS)
        if (kt + BK < K) {
            __builtin_prefetch(Ab + arow0 + BK, 0, 1);
            __builtin_prefetch(Bb + brow0 + BK, 0, 1);
        }
#endif
    };

    const int nk = K / BK;
    issue_tile(0, 0);

    for (int i = 0; i < nk; ++i) {
        const int buf = i & 1;
        // all waves done reading buf^1 (iter i-1) before it is overwritten
        __syncthreads();
        if (i + 1 < nk) {
            issue_tile((i + 1) * BK, buf ^ 1);
            wait_async_tail4();                 // tile i's 4 copies landed
        } else {
            wait_async_all();
        }
        __syncthreads();                        // every wave's copies visible

        // A fragment per ISA 16-bit 16x32 layout: lanes 0-15 hold K {0..7,16..23},
        // lanes 16-31 hold K {8..15,24..31} -> two contiguous 16B LDS reads
        v16bf afrag[2];
#pragma unroll
        for (int wm = 0; wm < 2; wm++) {
            const int m = wmb + wm * 16 + lrow;
            v8bf lo = *(const v8bf*)&sA[buf][m][khalf * 8];
            v8bf hi = *(const v8bf*)&sA[buf][m][16 + khalf * 8];
            afrag[wm] = __builtin_shufflevector(lo, hi, 0,1,2,3,4,5,6,7,8,9,10,11,12,13,14,15);
        }
        // B fragment: lane holds one N column, same K layout as A: {0..7,16..23} / {8..15,24..31}
        v16bf bfrag[4];
#pragma unroll
        for (int wn = 0; wn < 4; wn++) {
            const int n = wnb + wn * 16 + lrow;
            v8bf lo = *(const v8bf*)&sB[buf][n][khalf * 8];
            v8bf hi = *(const v8bf*)&sB[buf][n][16 + khalf * 8];
            bfrag[wn] = __builtin_shufflevector(lo, hi, 0,1,2,3,4,5,6,7,8,9,10,11,12,13,14,15);
        }

#pragma unroll
        for (int wm = 0; wm < 2; wm++)
#pragma unroll
            for (int wn = 0; wn < 4; wn++)
                acc[wm][wn] = __builtin_amdgcn_wmma_f32_16x16x32_bf16(
                    false, afrag[wm], false, bfrag[wn],
                    (short)0, acc[wm][wn], false, false);
    }

    // epilogue: C/D layout = VGPR r -> M=r (lanes 0-15) / M=r+8 (lanes 16-31), N=lane&15
    float*  Cf = (float*)C;
    bf16_t* Ch = (bf16_t*)C;
    const long cz = (long)z * c_zstride;
    const int mro = khalf * 8;
#pragma unroll
    for (int wm = 0; wm < 2; wm++) {
#pragma unroll
        for (int wn = 0; wn < 4; wn++) {
            const int gn = n0 + wnb + wn * 16 + lrow;
            const float bval = biasz ? biasz[gn] : 0.f;
#pragma unroll
            for (int r = 0; r < 8; r++) {
                const int gm = m0 + wmb + wm * 16 + mro + r;
                float v = acc[wm][wn][r] * alpha + bval;
                if (flags & EPI_RELU) v = v > 0.f ? v : 0.f;
                const long off = cz + (long)gm * ldc + gn;
                if (flags & EPI_BF16) Ch[off] = (bf16_t)v;
                else                  Cf[off] = v;
            }
        }
    }
}

// ---------------------------------------------------------------------------
// fp32 -> bf16 elementwise
// ---------------------------------------------------------------------------
__global__ void f32_to_bf16(const float* __restrict__ in, bf16_t* __restrict__ out, long n)
{
    long i = (long)blockIdx.x * blockDim.x + threadIdx.x;
    if (i < n) out[i] = (bf16_t)in[i];
}

// ---------------------------------------------------------------------------
// Batched transpose (+convert to bf16): in [z][R][C] -> out [z][C][R]
// ---------------------------------------------------------------------------
template <typename T>
__global__ void transpose_to_bf16(const T* __restrict__ in, bf16_t* __restrict__ out,
                                  int R, int C)
{
    __shared__ float tile[32][33];
    const long zoff = (long)blockIdx.z * (long)R * C;
    const int c0 = blockIdx.x * 32, r0 = blockIdx.y * 32;
    const int tx = threadIdx.x & 31, ty = threadIdx.x >> 5;  // 256 thr: 32x8
#pragma unroll
    for (int i = 0; i < 32; i += 8)
        tile[ty + i][tx] = (float)in[zoff + (long)(r0 + ty + i) * C + (c0 + tx)];
    __syncthreads();
#pragma unroll
    for (int i = 0; i < 32; i += 8)
        out[zoff + (long)(c0 + ty + i) * R + (r0 + tx)] = (bf16_t)tile[tx][ty + i];
}

// ---------------------------------------------------------------------------
// Causal softmax over rows of scores [H*S][S]; writes bf16 probs (0 for t>s)
// ---------------------------------------------------------------------------
__global__ __launch_bounds__(256)
void softmax_causal(const float* __restrict__ scores, bf16_t* __restrict__ attn)
{
    __shared__ float red[256];
    const long row = blockIdx.x;                 // h*S + s
    const int  s   = (int)(row & (S_DIM - 1));
    const int  L   = s + 1;                      // valid length (causal)
    const float* x = scores + row * (long)S_DIM;
    bf16_t*      y = attn   + row * (long)S_DIM;
    const int tid = threadIdx.x;

    float m = -INFINITY;
    for (int t = tid; t < L; t += 256) m = fmaxf(m, x[t]);
    red[tid] = m; __syncthreads();
    for (int o = 128; o > 0; o >>= 1) { if (tid < o) red[tid] = fmaxf(red[tid], red[tid + o]); __syncthreads(); }
    m = red[0]; __syncthreads();

    float sum = 0.f;
    for (int t = tid; t < L; t += 256) sum += __expf(x[t] - m);
    red[tid] = sum; __syncthreads();
    for (int o = 128; o > 0; o >>= 1) { if (tid < o) red[tid] += red[tid + o]; __syncthreads(); }
    const float inv = 1.f / red[0];

    for (int t = tid; t < S_DIM; t += 256)
        y[t] = (t < L) ? (bf16_t)(__expf(x[t] - m) * inv) : (bf16_t)0.f;
}

// ---------------------------------------------------------------------------
// y = a + b ; accumulate global sum / sumsq (for whole-tensor LayerNorm)
// ---------------------------------------------------------------------------
__global__ __launch_bounds__(256)
void add_reduce(const float* __restrict__ a, const float* __restrict__ b,
                float* __restrict__ y, float* __restrict__ sums, long n)
{
    __shared__ float r1[256], r2[256];
    const int tid = threadIdx.x;
    float s = 0.f, s2 = 0.f;
    for (long i = (long)blockIdx.x * 256 + tid; i < n; i += (long)gridDim.x * 256) {
        float v = a[i] + b[i];
        y[i] = v; s += v; s2 += v * v;
    }
    r1[tid] = s; r2[tid] = s2; __syncthreads();
    for (int o = 128; o > 0; o >>= 1) {
        if (tid < o) { r1[tid] += r1[tid + o]; r2[tid] += r2[tid + o]; }
        __syncthreads();
    }
    if (tid == 0) { atomicAdd(&sums[0], r1[0]); atomicAdd(&sums[1], r2[0]); }
}

// ---------------------------------------------------------------------------
// Apply global LN: out = (y - mean)*rsqrt(var+eps)*g + b  (f32 + optional bf16)
// ---------------------------------------------------------------------------
__global__ void ln_apply(const float* __restrict__ y, const float* __restrict__ sums,
                         const float* __restrict__ g, const float* __restrict__ b,
                         float* __restrict__ out32, bf16_t* __restrict__ out16, long n)
{
    const float mean = sums[0] / (float)n;
    const float var  = sums[1] / (float)n - mean * mean;
    const float rstd = rsqrtf(var + LN_EPS);
    long i = (long)blockIdx.x * blockDim.x + threadIdx.x;
    if (i < n) {
        float v = (y[i] - mean) * rstd * g[i] + b[i];
        out32[i] = v;
        if (out16) out16[i] = (bf16_t)v;
    }
}

__global__ void zero_floats(float* p, int n)
{
    if (threadIdx.x < n) p[threadIdx.x] = 0.f;
}

// ---------------------------------------------------------------------------
// Host orchestration
// ---------------------------------------------------------------------------
extern "C" void kernel_launch(void* const* d_in, const int* in_sizes, int n_in,
                              void* d_out, int out_size, void* d_ws, size_t ws_size,
                              hipStream_t stream)
{
    const float* x    = (const float*)d_in[0];
    const float* Wq   = (const float*)d_in[1];
    const float* bq   = (const float*)d_in[2];
    const float* Wk   = (const float*)d_in[3];
    const float* bk   = (const float*)d_in[4];
    const float* Wv   = (const float*)d_in[5];
    const float* bv   = (const float*)d_in[6];
    const float* ln_g = (const float*)d_in[7];
    const float* ln_b = (const float*)d_in[8];
    const float* W1   = (const float*)d_in[9];
    const float* b1   = (const float*)d_in[10];
    const float* W2   = (const float*)d_in[11];
    const float* b2   = (const float*)d_in[12];
    float* out = (float*)d_out;

    // workspace carve (256B aligned regions)
    char* p = (char*)d_ws;
    auto bump = [&](size_t elems, size_t esz) -> void* {
        void* r = (void*)p;
        p += (elems * esz + 255) & ~(size_t)255;
        return r;
    };
    const long SE = (long)S_DIM * E_DIM;
    bf16_t* x16   = (bf16_t*)bump(SE, 2);
    bf16_t* Wqt   = (bf16_t*)bump((long)H_DIM * KD_DIM * E_DIM, 2);
    bf16_t* Wkt   = (bf16_t*)bump((long)H_DIM * KD_DIM * E_DIM, 2);
    bf16_t* Wvt   = (bf16_t*)bump((long)H_DIM * E_DIM * E_DIM, 2);
    bf16_t* W1t   = (bf16_t*)bump((long)E_DIM * FF_DIM, 2);
    bf16_t* W2t   = (bf16_t*)bump((long)E_DIM * FF_DIM, 2);
    bf16_t* q16   = (bf16_t*)bump((long)H_DIM * S_DIM * KD_DIM, 2);
    bf16_t* k16   = (bf16_t*)bump((long)H_DIM * S_DIM * KD_DIM, 2);
    bf16_t* v16   = (bf16_t*)bump((long)H_DIM * S_DIM * E_DIM, 2);
    bf16_t* vt16  = (bf16_t*)bump((long)H_DIM * S_DIM * E_DIM, 2);
    float*  scr   = (float*) bump((long)H_DIM * S_DIM * S_DIM, 4);
    bf16_t* attn  = (bf16_t*)bump((long)H_DIM * S_DIM * S_DIM, 2);
    float*  aout  = (float*) bump(SE, 4);
    float*  y1    = (float*) bump(SE, 4);
    float*  hf    = (float*) bump(SE, 4);
    bf16_t* h16   = (bf16_t*)bump(SE, 2);
    bf16_t* ff1   = (bf16_t*)bump((long)S_DIM * FF_DIM, 2);
    float*  ff2   = (float*) bump(SE, 4);
    float*  y2    = (float*) bump(SE, 4);
    float*  sums  = (float*) bump(16, 4);

    const dim3 blk(256);

    // zero LN accumulators (ws is poisoned; must re-zero every call)
    zero_floats<<<dim3(1), dim3(32), 0, stream>>>(sums, 4);

    // ---- precision conversion / weight pre-transposition (bf16, [N][K]) ----
    f32_to_bf16<<<dim3((unsigned)(SE / 256)), blk, 0, stream>>>(x, x16, SE);
    transpose_to_bf16<float><<<dim3(KD_DIM/32, E_DIM/32, H_DIM), blk, 0, stream>>>(Wq, Wqt, E_DIM, KD_DIM);
    transpose_to_bf16<float><<<dim3(KD_DIM/32, E_DIM/32, H_DIM), blk, 0, stream>>>(Wk, Wkt, E_DIM, KD_DIM);
    transpose_to_bf16<float><<<dim3(E_DIM/32,  E_DIM/32, H_DIM), blk, 0, stream>>>(Wv, Wvt, E_DIM, E_DIM);
    transpose_to_bf16<float><<<dim3(FF_DIM/32, E_DIM/32, 1),     blk, 0, stream>>>(W1, W1t, E_DIM, FF_DIM);
    transpose_to_bf16<float><<<dim3(E_DIM/32,  FF_DIM/32, 1),    blk, 0, stream>>>(W2, W2t, FF_DIM, E_DIM);

    // ---- q/k/v projections (per-head, bf16 WMMA, f32 acc, bias epilogue) ----
    gemm_bf16_wmma<<<dim3(KD_DIM/BN, S_DIM/BM, H_DIM), blk, 0, stream>>>(
        x16, Wqt, q16, bq, S_DIM, KD_DIM, E_DIM,
        E_DIM, E_DIM, KD_DIM,
        0L, (long)KD_DIM * E_DIM, (long)S_DIM * KD_DIM, KD_DIM,
        11 /* log2(E) */, 0L, 0L, 1.0f, EPI_BF16);
    gemm_bf16_wmma<<<dim3(KD_DIM/BN, S_DIM/BM, H_DIM), blk, 0, stream>>>(
        x16, Wkt, k16, bk, S_DIM, KD_DIM, E_DIM,
        E_DIM, E_DIM, KD_DIM,
        0L, (long)KD_DIM * E_DIM, (long)S_DIM * KD_DIM, KD_DIM,
        11, 0L, 0L, 1.0f, EPI_BF16);
    gemm_bf16_wmma<<<dim3(E_DIM/BN, S_DIM/BM, H_DIM), blk, 0, stream>>>(
        x16, Wvt, v16, bv, S_DIM, E_DIM, E_DIM,
        E_DIM, E_DIM, E_DIM,
        0L, (long)E_DIM * E_DIM, (long)S_DIM * E_DIM, E_DIM,
        11, 0L, 0L, 1.0f, EPI_BF16);

    // v^T per head: [H][S][E] -> [H][E][S] so attn@v uses contiguous B-tiles
    transpose_to_bf16<bf16_t><<<dim3(E_DIM/32, S_DIM/32, H_DIM), blk, 0, stream>>>(v16, vt16, S_DIM, E_DIM);

    // ---- scores = q k^T / sqrt(KD), causal blocks above diagonal skipped ----
    gemm_bf16_wmma<<<dim3(S_DIM/BN, S_DIM/BM, H_DIM), blk, 0, stream>>>(
        q16, k16, scr, nullptr, S_DIM, S_DIM, KD_DIM,
        KD_DIM, KD_DIM, S_DIM,
        (long)S_DIM * KD_DIM, (long)S_DIM * KD_DIM, (long)S_DIM * S_DIM, 0L,
        7 /* log2(KD) */, 0L, 0L, 0.0883883476483f /* 1/sqrt(128) */, EPI_CAUSAL);

    softmax_causal<<<dim3(H_DIM * S_DIM), blk, 0, stream>>>(scr, attn);

    // ---- attn_out[s,f] = sum_{h,t} attn[h,s,t] v[h,t,f] : single GEMM,
    //      K = H*S with per-head super-batch strides on A and B -------------
    gemm_bf16_wmma<<<dim3(E_DIM/BN, S_DIM/BM, 1), blk, 0, stream>>>(
        attn, vt16, aout, nullptr, S_DIM, E_DIM, H_DIM * S_DIM,
        S_DIM, S_DIM, E_DIM,
        0L, 0L, 0L, 0L,
        11 /* log2(S) */, (long)S_DIM * S_DIM, (long)E_DIM * S_DIM, 1.0f, 0);

    // ---- h = LN(x + attn_out)  (global mean/var over the whole [S,E]) ----
    add_reduce<<<dim3(2048), blk, 0, stream>>>(x, aout, y1, &sums[0], SE);
    ln_apply<<<dim3((unsigned)(SE / 256)), blk, 0, stream>>>(y1, &sums[0], ln_g, ln_b, hf, h16, SE);

    // ---- FFN: relu(h W1 + b1) W2 + b2 ----
    gemm_bf16_wmma<<<dim3(FF_DIM/BN, S_DIM/BM, 1), blk, 0, stream>>>(
        h16, W1t, ff1, b1, S_DIM, FF_DIM, E_DIM,
        E_DIM, E_DIM, FF_DIM,
        0L, 0L, 0L, 0L,
        11, 0L, 0L, 1.0f, EPI_RELU | EPI_BF16);
    gemm_bf16_wmma<<<dim3(E_DIM/BN, S_DIM/BM, 1), blk, 0, stream>>>(
        ff1, W2t, ff2, b2, S_DIM, E_DIM, FF_DIM,
        FF_DIM, FF_DIM, E_DIM,
        0L, 0L, 0L, 0L,
        13 /* log2(FF) */, 0L, 0L, 1.0f, 0);

    // ---- out = LN(h + ff) ----
    add_reduce<<<dim3(2048), blk, 0, stream>>>(hf, ff2, y2, &sums[2], SE);
    ln_apply<<<dim3((unsigned)(SE / 256)), blk, 0, stream>>>(y2, &sums[2], ln_g, ln_b, out, nullptr, SE);
}
